// EncoderDecoderAttn_26920855011596
// MI455X (gfx1250) — matrix operand order repaired and checked by reading, not compiled, not run on hardware
//
#include <hip/hip_runtime.h>
#include <math.h>

#define SLEN 4096
#define NB   32
#define EDIM 256
#define DDIM 256
#define ADIM 256

typedef __attribute__((ext_vector_type(16))) __bf16         v16bf;
typedef __attribute__((ext_vector_type(8)))  float          v8f;
typedef __attribute__((ext_vector_type(16))) unsigned short v16u;

__device__ __forceinline__ unsigned short f32_to_bf16_rne(float f) {
  union { float f; unsigned u; } v; v.f = f;
  unsigned u = v.u;
  u += 0x7FFFu + ((u >> 16) & 1u);   // round-to-nearest-even
  return (unsigned short)(u >> 16);
}

// Native v_tanh_f32 on gfx1250 (TRANS op); branch-free fallback otherwise.
__device__ __forceinline__ float fast_tanhf(float x) {
#if __has_builtin(__builtin_amdgcn_tanhf)
  return __builtin_amdgcn_tanhf(x);
#elif __has_builtin(__builtin_amdgcn_tanh_f32)
  return __builtin_amdgcn_tanh_f32(x);
#else
  // tanh(x) = sign(x) * (1 - 2/(exp(2|x|)+1)); overflow -> exp=inf -> t=1.
  float ax = fabsf(x);
  float e  = __expf(2.0f * ax);
  float t  = 1.0f - __fdividef(2.0f, e + 1.0f);
  return copysignf(t, x);
#endif
}

// ---------------------------------------------------------------------------
// Kernel 1: proj = dh @ Wd^T + bias  (32x256), and convert We -> bf16 bits.
// ---------------------------------------------------------------------------
__global__ __launch_bounds__(256) void prep_kernel(
    const float* __restrict__ dh,        // (NB, DDIM)
    const float* __restrict__ W,         // (ADIM, DDIM+EDIM) row-major
    const float* __restrict__ bias,      // (ADIM)
    float* __restrict__ proj,            // (NB, ADIM)
    unsigned short* __restrict__ We16)   // (ADIM, EDIM) bf16 bits, row-major
{
  const int b = blockIdx.x;
  const int a = threadIdx.x;
  const float* dhb = dh + b * DDIM;
  const float* wa  = W + (size_t)a * (DDIM + EDIM);
  float acc = 0.f;
  for (int k = 0; k < DDIM; ++k) acc += dhb[k] * wa[k];
  proj[b * ADIM + a] = acc + bias[a];

  // 8192 threads total; each converts 8 elements of We (65536 total)
  const int t = blockIdx.x * 256 + threadIdx.x;
#pragma unroll
  for (int j = 0; j < 8; ++j) {
    int idx = t * 8 + j;
    int ar = idx >> 8, e = idx & 255;
    We16[idx] = f32_to_bf16_rne(W[(size_t)ar * (DDIM + EDIM) + DDIM + e]);
  }
}

// ---------------------------------------------------------------------------
// Kernel 2: WMMA scores pass.
// Per wave: TWO 16-row s-tiles (32 rows) of batch b. Both A-fragment sets
// (eo rows, f32->bf16) live in registers for the full K=256; every B
// fragment of We is loaded once and feeds two WMMAs (halves L2 traffic).
// score[s] = sum_a tanh(proj[b,a] + (eo @ We^T)[s,a]); also row L2 norms.
// ---------------------------------------------------------------------------
__global__ __launch_bounds__(256, 1) void scores_kernel(
    const float* __restrict__ eo,            // (SLEN, NB, EDIM)
    const float* __restrict__ proj,          // (NB, ADIM)
    const unsigned short* __restrict__ We16, // (ADIM, EDIM) bf16 bits
    float* __restrict__ scores,              // (NB, SLEN)
    float* __restrict__ norms)               // (NB, SLEN)
{
  const int lane = threadIdx.x & 31;
  const int wave = threadIdx.x >> 5;
  const int m    = lane & 15;   // row within tile (A) / column (B,C)
  const int hi   = lane >> 4;   // lane half selects K sub-range
  const int b    = blockIdx.y;
  const int s0   = blockIdx.x * 256 + wave * 32;   // 32 rows per wave

  const float* rowp0 = eo + ((size_t)(s0 + m)      * NB + b) * EDIM;
  const float* rowp1 = eo + ((size_t)(s0 + 16 + m) * NB + b) * EDIM;

  // Build 2x8 A-fragments (K = 256 in steps of 32), ISA 16-bit A layout:
  // lanes 0-15: K = k..k+7 (elems 0-7), k+16..k+23 (elems 8-15); lanes16-31:+8
  v16bf afrag0[8], afrag1[8];
  float sumsq0 = 0.f, sumsq1 = 0.f;
#pragma unroll
  for (int ks = 0; ks < 8; ++ks) {
    const int c0 = ks * 32 + hi * 8;
    v8f x0 = *(const v8f*)(rowp0 + c0);
    v8f x1 = *(const v8f*)(rowp0 + c0 + 16);
    v8f y0 = *(const v8f*)(rowp1 + c0);
    v8f y1 = *(const v8f*)(rowp1 + c0 + 16);
    v16u ra, rb;
#pragma unroll
    for (int j = 0; j < 8; ++j) {
      sumsq0 += x0[j] * x0[j] + x1[j] * x1[j];
      sumsq1 += y0[j] * y0[j] + y1[j] * y1[j];
      ra[j]     = f32_to_bf16_rne(x0[j]);
      ra[j + 8] = f32_to_bf16_rne(x1[j]);
      rb[j]     = f32_to_bf16_rne(y0[j]);
      rb[j + 8] = f32_to_bf16_rne(y1[j]);
    }
    afrag0[ks] = __builtin_bit_cast(v16bf, ra);
    afrag1[ks] = __builtin_bit_cast(v16bf, rb);
  }

  // Row norms: lanes m and m+16 together cover all 256 columns of a row.
  sumsq0 += __shfl_xor(sumsq0, 16, 32);
  sumsq1 += __shfl_xor(sumsq1, 16, 32);
  if (hi == 0) {
    norms[(size_t)b * SLEN + s0 + m]      = sqrtf(sumsq0);
    norms[(size_t)b * SLEN + s0 + 16 + m] = sqrtf(sumsq1);
  }

  float acc0[8], acc1[8];
#pragma unroll
  for (int r = 0; r < 8; ++r) { acc0[r] = 0.f; acc1[r] = 0.f; }

  // Loop over 16 N-tiles of the A-dim (a = nt*16 + lane&15 per C layout).
  for (int nt = 0; nt < 16; ++nt) {
    v8f c0 = {}, c1 = {};
    // B 32x16 bf16 layout: lane = N column, same K order as A:
    // lanes 0-15: K = k..k+7, k+16..k+23; lanes 16-31: +8.
    const unsigned short* wrow =
        We16 + (size_t)(nt * 16 + m) * EDIM + hi * 8;
#pragma unroll
    for (int ks = 0; ks < 8; ++ks) {
      typedef __attribute__((ext_vector_type(8))) unsigned short v8u;
      v8u blo = *(const v8u*)(wrow + ks * 32);
      v8u bhi = *(const v8u*)(wrow + ks * 32 + 16);
      v16u braw = __builtin_shufflevector(blo, bhi, 0, 1, 2, 3, 4, 5, 6, 7,
                                          8, 9, 10, 11, 12, 13, 14, 15);
      v16bf bf  = __builtin_bit_cast(v16bf, braw);
      c0 = __builtin_amdgcn_wmma_f32_16x16x32_bf16(
               false, afrag0[ks], false, bf, (short)0, c0, false, false);
      c1 = __builtin_amdgcn_wmma_f32_16x16x32_bf16(
               false, afrag1[ks], false, bf, (short)0, c1, false, false);
    }
    // C layout: VGPR r holds (M = r or r+8, N = nt*16 + (lane&15)).
    const float pj = proj[b * ADIM + nt * 16 + m];
#pragma unroll
    for (int r = 0; r < 8; ++r) {
      acc0[r] += fast_tanhf(c0[r] + pj);
      acc1[r] += fast_tanhf(c1[r] + pj);
    }
  }

  // Reduce over N (16-lane halves): xor 1,2,4,8 stays within each half.
#pragma unroll
  for (int r = 0; r < 8; ++r) {
#pragma unroll
    for (int mask = 1; mask < 16; mask <<= 1) {
      acc0[r] += __shfl_xor(acc0[r], mask, 32);
      acc1[r] += __shfl_xor(acc1[r], mask, 32);
    }
  }
  if (lane == 0) {
#pragma unroll
    for (int r = 0; r < 8; ++r) {
      scores[(size_t)b * SLEN + s0 + r]      = acc0[r];
      scores[(size_t)b * SLEN + s0 + 16 + r] = acc1[r];
    }
  } else if (lane == 16) {
#pragma unroll
    for (int r = 0; r < 8; ++r) {
      scores[(size_t)b * SLEN + s0 + 8 + r]  = acc0[r];
      scores[(size_t)b * SLEN + s0 + 24 + r] = acc1[r];
    }
  }
}

// ---------------------------------------------------------------------------
// Kernel 3: softmax over S per batch (in-place on scores) + attn_map to d_out.
// ---------------------------------------------------------------------------
__global__ __launch_bounds__(256) void softmax_kernel(
    float* __restrict__ scores,       // (NB, SLEN): in scores, out softmax p
    const float* __restrict__ norms,  // (NB, SLEN)
    float* __restrict__ out)          // d_out: [0,8192)=attended, then attn_map
{
  const int b   = blockIdx.x;
  const int tid = threadIdx.x;
  __shared__ float red[256];
  float* srow = scores + (size_t)b * SLEN;

  float lmax = -3.402823466e+38f;
  for (int s = tid; s < SLEN; s += 256) lmax = fmaxf(lmax, srow[s]);
  red[tid] = lmax;
  __syncthreads();
  for (int off = 128; off > 0; off >>= 1) {
    if (tid < off) red[tid] = fmaxf(red[tid], red[tid + off]);
    __syncthreads();
  }
  const float mx = red[0];
  __syncthreads();

  float lsum = 0.f;
  for (int s = tid; s < SLEN; s += 256) {
    float e = __expf(srow[s] - mx);
    srow[s] = e;
    lsum += e;
  }
  red[tid] = lsum;
  __syncthreads();
  for (int off = 128; off > 0; off >>= 1) {
    if (tid < off) red[tid] += red[tid + off];
    __syncthreads();
  }
  const float inv = 1.0f / red[0];

  const float* nrow = norms + (size_t)b * SLEN;
  float* amap = out + NB * EDIM + (size_t)b * SLEN;
  for (int s = tid; s < SLEN; s += 256) {
    float p = srow[s] * inv;
    srow[s] = p;
    amap[s] = p * nrow[s];
  }
}

// ---------------------------------------------------------------------------
// Kernel 4: attended partials: for each (b, s-chunk) accumulate p[s]*eo[b,s,e].
// Deterministic (no atomics); fixed-order reduce in kernel 5.
// ---------------------------------------------------------------------------
__global__ __launch_bounds__(256) void attend_partial_kernel(
    const float* __restrict__ eo,      // (SLEN, NB, EDIM)
    const float* __restrict__ p,       // (NB, SLEN) softmax
    float* __restrict__ partials)      // (NB, 16, EDIM)
{
  const int b  = blockIdx.x;
  const int ch = blockIdx.y;
  const int e  = threadIdx.x;
  const float* prow = p + (size_t)b * SLEN + ch * 256;
  float acc = 0.f;
  for (int i = 0; i < 256; ++i) {
    const int s = ch * 256 + i;
    acc += prow[i] * eo[((size_t)s * NB + b) * EDIM + e];
  }
  partials[((size_t)b * 16 + ch) * EDIM + e] = acc;
}

__global__ __launch_bounds__(256) void attend_reduce_kernel(
    const float* __restrict__ partials, float* __restrict__ out)
{
  const int b = blockIdx.x;
  const int e = threadIdx.x;
  float acc = 0.f;
  for (int ch = 0; ch < 16; ++ch)
    acc += partials[((size_t)b * 16 + ch) * EDIM + e];
  out[b * EDIM + e] = acc;
}

// ---------------------------------------------------------------------------
extern "C" void kernel_launch(void* const* d_in, const int* in_sizes, int n_in,
                              void* d_out, int out_size, void* d_ws, size_t ws_size,
                              hipStream_t stream) {
  (void)in_sizes; (void)n_in; (void)out_size; (void)ws_size;
  const float* dh   = (const float*)d_in[0];  // (1, NB, DDIM)
  const float* eo   = (const float*)d_in[1];  // (SLEN, NB, EDIM)
  const float* W    = (const float*)d_in[2];  // (ADIM, DDIM+EDIM)
  const float* bias = (const float*)d_in[3];  // (ADIM)
  float* out = (float*)d_out;

  char* ws = (char*)d_ws;
  float*          proj     = (float*)(ws);                  //  32 KB
  unsigned short* We16     = (unsigned short*)(ws + 32768); // 128 KB
  float*          scores   = (float*)(ws + 163840);         // 512 KB
  float*          norms    = (float*)(ws + 688128);         // 512 KB
  float*          partials = (float*)(ws + 1212416);        // 512 KB

  prep_kernel<<<NB, 256, 0, stream>>>(dh, W, bias, proj, We16);
  scores_kernel<<<dim3(SLEN / 256, NB), 256, 0, stream>>>(eo, proj, We16,
                                                          scores, norms);
  softmax_kernel<<<NB, 256, 0, stream>>>(scores, norms, out);
  attend_partial_kernel<<<dim3(NB, 16), 256, 0, stream>>>(eo, scores, partials);
  attend_reduce_kernel<<<NB, 256, 0, stream>>>(partials, out);
}
